// DecoderNoTrainer_9594956939690
// MI455X (gfx1250) — matrix operand order repaired and checked by reading, not compiled, not run on hardware
//
#include <hip/hip_runtime.h>
#include <hip/hip_bf16.h>

// ---------------------------------------------------------------------------
// Problem dims (from reference setup_inputs):
//   B=512, L=128, D=512, E=256, V=1024, gates 4D=2048
// ---------------------------------------------------------------------------
#define BN   512
#define LN   128
#define DN   512
#define EN   256
#define VN   1024
#define G4D  (4 * DN)
#define BIG_NUMBER 1.0e6f
#define SOS_CODE 1

typedef __attribute__((ext_vector_type(16))) __bf16 v16bf;
typedef __attribute__((ext_vector_type(8)))  float  v8f;

// ---------------------------------------------------------------------------
// Pack a row-major fp32 weight W[K,N] into the CDNA5 WMMA B-matrix lane layout
// (bf16).  Tile = 32(K) x 16(N).  Storage: [nTile][kTile][lane(32)][e(16)],
// element value = W[kTile*32 + 8*(lane/16) + e%8 + 16*(e/8)][nTile*16 + lane%16].
// Each lane's 16 bf16 (32B) are contiguous -> per-step loads are 2x b128.
// ---------------------------------------------------------------------------
__global__ __launch_bounds__(256) void pack_b_kernel(
    const float* __restrict__ W, __bf16* __restrict__ Bp, int K, int N) {
  int t = blockIdx.x * blockDim.x + threadIdx.x;
  int total = K * N;
  if (t >= total) return;
  int Kt   = K >> 5;
  int e    = t & 15;
  int lane = (t >> 4) & 31;
  int tile = t >> 9;
  int kt   = tile % Kt;
  int nt   = tile / Kt;
  int k = kt * 32 + ((lane >> 4) << 3) + (e & 7) + ((e >> 3) << 4);
  int n = nt * 16 + (lane & 15);
  Bp[t] = (__bf16)W[(size_t)k * N + n];
}

// ---------------------------------------------------------------------------
// Init per-call state: h=dec_h0, c=dec_c0, mask=0, tok=SOS.
// ---------------------------------------------------------------------------
__global__ __launch_bounds__(256) void init_state_kernel(
    const float* __restrict__ h0, const float* __restrict__ c0,
    float* __restrict__ h, float* __restrict__ c,
    float* __restrict__ mask, int* __restrict__ tok) {
  int t = blockIdx.x * blockDim.x + threadIdx.x;
  if (t < BN * DN) { h[t] = h0[t]; c[t] = c0[t]; }
  if (t < BN * LN) mask[t] = 0.0f;
  if (t < BN) tok[t] = SOS_CODE;
}

// ---------------------------------------------------------------------------
// Per-lane A row base pointer (row-gather resolved ONCE, outside the K-loop).
// Lane l covers row M = mt*16 + l%16, starting at float offset 8*(l/16).
// ---------------------------------------------------------------------------
__device__ __forceinline__ const float* a_row_ptr(
    const float* __restrict__ A, const int* __restrict__ idx, int lda,
    int mt, int lane) {
  int m   = mt * 16 + (lane & 15);
  int row = idx ? idx[m] : m;
  return A + (size_t)row * lda + ((lane >> 4) << 3);
}

// A-tile fragment kt for lane: two contiguous 8-float groups at immediate
// offsets kt*32 and kt*32+16 from the per-lane row base, converted to bf16.
// (Layout: lane l, element e -> K = (e%8) + 8*(l/16) + 16*(e/8).)
__device__ __forceinline__ v16bf load_a_tile(const float* __restrict__ p,
                                             int kt) {
  v8f g0 = *(const v8f*)(p + kt * 32);
  v8f g1 = *(const v8f*)(p + kt * 32 + 16);
  v16bf a;
#pragma unroll
  for (int e = 0; e < 8; ++e) {
    a[e]     = (__bf16)g0[e];
    a[e + 8] = (__bf16)g1[e];
  }
  return a;
}

__device__ __forceinline__ v16bf load_b_tile(
    const __bf16* __restrict__ Bp, int Kt, int nt, int kt, int lane) {
  return *(const v16bf*)(Bp + ((((size_t)nt * Kt + kt) * 32 + lane) << 4));
}

// ---------------------------------------------------------------------------
// Fully-unrolled, software-pipelined K-loop for one A/B pair (K is a
// compile-time constant: 256 or 512 here).  The prefetch of iteration kt+1 is
// INTERLEAVED with the 4 WMMAs of kt (load bn[j]; wmma b[j]) so peak live B
// registers stay low while every WMMA operand was fetched a full iteration
// (>= 4 WMMAs) earlier.  Full unrolling lets SSA renaming kill the rotation.
// ---------------------------------------------------------------------------
template <int K>
__device__ __forceinline__ void gemm_pair(
    const float* __restrict__ Ap,        // per-lane row base (gather resolved)
    const __bf16* __restrict__ Bp, const int nt[4], int lane, v8f acc[4]) {
  constexpr int Kt = K >> 5;
  v16bf a = load_a_tile(Ap, 0);
  v16bf b[4];
#pragma unroll
  for (int j = 0; j < 4; ++j) b[j] = load_b_tile(Bp, Kt, nt[j], 0, lane);
#pragma unroll
  for (int kt = 0; kt < Kt - 1; ++kt) {
    v16bf an = load_a_tile(Ap, kt + 1);
    v16bf bn[4];
#pragma unroll
    for (int j = 0; j < 4; ++j) {
      bn[j] = load_b_tile(Bp, Kt, nt[j], kt + 1, lane);
      acc[j] = __builtin_amdgcn_wmma_f32_16x16x32_bf16(
          false, a, false, b[j], (short)0, acc[j], false, false);
    }
    a = an;
#pragma unroll
    for (int j = 0; j < 4; ++j) b[j] = bn[j];
  }
#pragma unroll
  for (int j = 0; j < 4; ++j)
    acc[j] = __builtin_amdgcn_wmma_f32_16x16x32_bf16(
        false, a, false, b[j], (short)0, acc[j], false, false);
}

__device__ __forceinline__ float sigmoidf_fast(float x) {
  return 1.0f / (1.0f + __expf(-x));
}

// ---------------------------------------------------------------------------
// Fused LSTM-cell GEMM:  z = emb[tok]@Wx + h@Wh + b, gates applied in the
// epilogue.  Wave computes the SAME 16x16 (rows x d) tile for all four gates
// (i,f,g,o), i.e. columns {g*D + dtile*16} of z, then updates c,h in place.
// Grid: (B/16)*(D/16) waves.
// ---------------------------------------------------------------------------
__global__ __launch_bounds__(256) void lstm_fused_gemm_kernel(
    const float* __restrict__ emb, const int* __restrict__ tok,
    const __bf16* __restrict__ Wxp,
    const float* __restrict__ hin, const __bf16* __restrict__ Whp,
    const float* __restrict__ bb,
    float* __restrict__ h, float* __restrict__ c) {
  int wave = (blockIdx.x * blockDim.x + threadIdx.x) >> 5;
  int lane = threadIdx.x & 31;
  const int dblk = DN / 16;              // 32 d-tiles
  int mt = wave / dblk;
  int dt = wave % dblk;
  if (mt * 16 >= BN) return;             // wave-uniform

  // gate g lives at z columns [g*DN + dt*16, +16) -> B n-tile = g*dblk + dt
  int nt[4] = {0 * dblk + dt, 1 * dblk + dt, 2 * dblk + dt, 3 * dblk + dt};

  v8f acc[4];
#pragma unroll
  for (int j = 0; j < 4; ++j) acc[j] = v8f{};

  const float* Apx = a_row_ptr(emb, tok, EN, mt, lane);   // gather once
  const float* Aph = a_row_ptr(hin, nullptr, DN, mt, lane);
  gemm_pair<EN>(Apx, Wxp, nt, lane, acc);                  // x @ Wx
  gemm_pair<DN>(Aph, Whp, nt, lane, acc);                  // h @ Wh

  // epilogue: LSTM cell update.  element r of lane l -> row = r + 8*(l/16).
  int colb   = lane & 15;
  int rowoff = (lane >> 4) << 3;
  int d  = dt * 16 + colb;
  float bi = bb[d], bf = bb[DN + d], bg = bb[2 * DN + d], bo = bb[3 * DN + d];
#pragma unroll
  for (int r = 0; r < 8; ++r) {
    int row = mt * 16 + rowoff + r;
    size_t off = (size_t)row * DN + d;
    float ig = sigmoidf_fast(acc[0][r] + bi);
    float fg = sigmoidf_fast(acc[1][r] + bf);
    float gg = acc[2][r] + bg;
    float og = sigmoidf_fast(acc[3][r] + bo);
    float cn = fg * c[off] + ig * tanhf(gg);
    c[off] = cn;
    h[off] = og * tanhf(cn);
  }
}

// ---------------------------------------------------------------------------
// Generic bf16-WMMA GEMM (K compile-time):  C[M,N] = A0[M,K]@B0 + bias.
// One wave computes a 16x64 strip.  Grid supplies exactly (M/16)*(N/64) waves.
// ---------------------------------------------------------------------------
template <int K>
__global__ __launch_bounds__(256) void gemm_bf16_wmma_kernel(
    const float* __restrict__ A0, int lda0,
    const __bf16* __restrict__ B0p,
    const float* __restrict__ bias, float* __restrict__ C, int M, int N) {
  int wave = (blockIdx.x * blockDim.x + threadIdx.x) >> 5;
  int lane = threadIdx.x & 31;
  int nblk = N >> 6;
  int mt   = wave / nblk;
  int nt0  = (wave % nblk) << 2;
  if (mt * 16 >= M) return;

  int nt[4] = {nt0, nt0 + 1, nt0 + 2, nt0 + 3};
  v8f acc[4];
#pragma unroll
  for (int j = 0; j < 4; ++j) acc[j] = v8f{};

  const float* Ap = a_row_ptr(A0, nullptr, lda0, mt, lane);
  gemm_pair<K>(Ap, B0p, nt, lane, acc);

  int colb   = lane & 15;
  int rowoff = (lane >> 4) << 3;
#pragma unroll
  for (int j = 0; j < 4; ++j) {
    int n = nt[j] * 16 + colb;
    float bv = bias ? bias[n] : 0.0f;
#pragma unroll
    for (int r = 0; r < 8; ++r) {
      C[(size_t)(mt * 16 + rowoff + r) * N + n] = acc[j][r] + bv;
    }
  }
}

// ---------------------------------------------------------------------------
// Attention step: per batch row b (one block, 128 threads = L):
//   logit[l] = tanh(enc_proj[b,l,:] + q[b,:]) . v + bv - mask*BIG
//   pointers = softmax(logit); idx = argmax (first-max); update mask, tok.
// ---------------------------------------------------------------------------
__global__ __launch_bounds__(128) void attn_step_kernel(
    const float* __restrict__ ep, const float* __restrict__ q,
    const float* __restrict__ v, const float* __restrict__ bvp,
    float* __restrict__ mask, int* __restrict__ tok,
    const int* __restrict__ enc_input, float* __restrict__ out, int step) {
  int b = blockIdx.x;
  int t = threadIdx.x;
  __shared__ float qs[DN];
  __shared__ float vs[DN];
  __shared__ float red[LN];
  __shared__ int   redi[LN];

  for (int d = t; d < DN; d += LN) {
    qs[d] = q[(size_t)b * DN + d];
    vs[d] = v[d];
  }
  __syncthreads();

  const float* eprow = ep + ((size_t)b * LN + t) * DN;
  float acc = 0.0f;
  for (int d0 = 0; d0 < DN; d0 += 8) {
    v8f e = *(const v8f*)(eprow + d0);
#pragma unroll
    for (int k = 0; k < 8; ++k) acc += tanhf(e[k] + qs[d0 + k]) * vs[d0 + k];
  }
  float logit = acc + bvp[0] - mask[(size_t)b * LN + t] * BIG_NUMBER;

  // block max + first-index argmax
  red[t] = logit; redi[t] = t;
  __syncthreads();
  for (int s = LN >> 1; s > 0; s >>= 1) {
    if (t < s) {
      float o = red[t + s]; int oi = redi[t + s];
      if (o > red[t] || (o == red[t] && oi < redi[t])) { red[t] = o; redi[t] = oi; }
    }
    __syncthreads();
  }
  float maxv = red[0];
  int   maxi = redi[0];
  __syncthreads();

  float p = __expf(logit - maxv);
  red[t] = p;
  __syncthreads();
  for (int s = LN >> 1; s > 0; s >>= 1) {
    if (t < s) red[t] += red[t + s];
    __syncthreads();
  }
  out[((size_t)b * LN + step) * LN + t] = p / red[0];

  if (t == 0) {
    mask[(size_t)b * LN + maxi] += 1.0f;
    tok[b] = enc_input[(size_t)b * LN + maxi];
  }
}

// ---------------------------------------------------------------------------
// Host launcher
// ---------------------------------------------------------------------------
extern "C" void kernel_launch(void* const* d_in, const int* in_sizes, int n_in,
                              void* d_out, int out_size, void* d_ws, size_t ws_size,
                              hipStream_t stream) {
  (void)in_sizes; (void)n_in; (void)out_size; (void)ws_size;
  const float* dec_h0    = (const float*)d_in[0];
  const float* dec_c0    = (const float*)d_in[1];
  const float* enc_out   = (const float*)d_in[2];   // [B,L,D]
  const int*   enc_input = (const int*)  d_in[3];   // [B,L]
  const float* emb       = (const float*)d_in[4];   // [V,E]
  const float* Wx        = (const float*)d_in[5];   // [E,4D]
  const float* Wh        = (const float*)d_in[6];   // [D,4D]
  const float* bb        = (const float*)d_in[7];   // [4D]
  const float* W1        = (const float*)d_in[8];   // [D,D]
  const float* b1        = (const float*)d_in[9];   // [D]
  const float* W2        = (const float*)d_in[10];  // [D,D]
  const float* b2        = (const float*)d_in[11];  // [D]
  const float* vvec      = (const float*)d_in[12];  // [D]
  const float* bv        = (const float*)d_in[13];  // scalar
  float* out = (float*)d_out;                       // [B,L,L]

  // workspace carve-up (256B aligned)
  char* ws = (char*)d_ws;
  size_t o = 0;
  auto take = [&](size_t bytes) { char* p = ws + o; o += (bytes + 255) & ~(size_t)255; return p; };
  __bf16* Wxp = (__bf16*)take((size_t)EN * G4D * 2);
  __bf16* Whp = (__bf16*)take((size_t)DN * G4D * 2);
  __bf16* W1p = (__bf16*)take((size_t)DN * DN * 2);
  __bf16* W2p = (__bf16*)take((size_t)DN * DN * 2);
  float* enc_proj = (float*)take((size_t)BN * LN * DN * 4);  // 128 MB
  float* q   = (float*)take((size_t)BN * DN * 4);
  float* h   = (float*)take((size_t)BN * DN * 4);
  float* c   = (float*)take((size_t)BN * DN * 4);
  float* mask = (float*)take((size_t)BN * LN * 4);
  int*   tok  = (int*)take((size_t)BN * 4);

  // 1) pack weights into WMMA B layout
  auto packGrid = [](int n) { return dim3((n + 255) / 256); };
  pack_b_kernel<<<packGrid(EN * G4D), 256, 0, stream>>>(Wx, Wxp, EN, G4D);
  pack_b_kernel<<<packGrid(DN * G4D), 256, 0, stream>>>(Wh, Whp, DN, G4D);
  pack_b_kernel<<<packGrid(DN * DN),  256, 0, stream>>>(W1, W1p, DN, DN);
  pack_b_kernel<<<packGrid(DN * DN),  256, 0, stream>>>(W2, W2p, DN, DN);

  // 2) init state
  init_state_kernel<<<packGrid(BN * DN), 256, 0, stream>>>(dec_h0, dec_c0, h, c, mask, tok);

  // 3) enc_proj = enc_outputs @ W1 + b1   ([B*L, D])
  {
    int M = BN * LN, N = DN;
    int waves = (M / 16) * (N / 64);
    gemm_bf16_wmma_kernel<DN><<<dim3(waves / 8), 256, 0, stream>>>(
        enc_out, DN, W1p, b1, enc_proj, M, N);
  }

  // 4) decode loop
  for (int step = 0; step < LN; ++step) {
    // fused: z = emb[tok]@Wx + h@Wh + b; gates -> h,c
    {
      int waves = (BN / 16) * (DN / 16);
      lstm_fused_gemm_kernel<<<dim3(waves / 8), 256, 0, stream>>>(
          emb, tok, Wxp, h, Whp, bb, h, c);
    }
    // q = h @ W2 + b2
    {
      int waves = (BN / 16) * (DN / 64);
      gemm_bf16_wmma_kernel<DN><<<dim3(waves / 8), 256, 0, stream>>>(
          h, DN, W2p, b2, q, BN, DN);
    }
    attn_step_kernel<<<dim3(BN), LN, 0, stream>>>(
        enc_proj, q, vvec, bv, mask, tok, enc_input, out, step);
  }
}
